// BigBirdBlockSparseAttention_22101901705423
// MI455X (gfx1250) — matrix operand order repaired and checked by reading, not compiled, not run on hardware
//
#include <hip/hip_runtime.h>

// BigBird block-sparse attention for MI455X (gfx1250, wave32, WMMA).
// Phase 0: elementwise f32->bf16 conversion of inputs/weights (once).
// Phase 1: QKV projections as pure-bf16 WMMA GEMMs -> blocked q/k/v in d_ws.
// Phase 2: flash-attention over per-(b,h,block) key-block lists;
//          K/V tiles staged once per workgroup into LDS (coalesced b128),
//          v_wmma_f32_16x16x32_bf16 for QK^T and P*V, fp32 softmax.

#define B_   2
#define S_   4096
#define H_   1024
#define NH_  16
#define HD_  64
#define BS_  64
#define L_   64
#define R_   3
#define PEN_ (-10000.0f)
#define LDP_ 72   // LDS row pitch (ushorts) for K/V tiles: bank-spread + 16B align

typedef __attribute__((ext_vector_type(16))) __bf16          v16bf;
typedef __attribute__((ext_vector_type(16))) unsigned short  v16u;
typedef __attribute__((ext_vector_type(8)))  float           v8f;

// Native f32->bf16 (backend selects hardware cvt on gfx1250; RNE).
__device__ __forceinline__ unsigned short f2bf(float f) {
  __bf16 h = (__bf16)f;
  return __builtin_bit_cast(unsigned short, h);
}
__device__ __forceinline__ unsigned int pk2bf(float lo, float hi) {
  return (unsigned int)f2bf(lo) | ((unsigned int)f2bf(hi) << 16);
}

__device__ __forceinline__ v8f wmma_bf16(v16u a, v16u b, v8f c) {
  return __builtin_amdgcn_wmma_f32_16x16x32_bf16(
      false, __builtin_bit_cast(v16bf, a),
      false, __builtin_bit_cast(v16bf, b),
      (short)0, c, false, false);
}

// ---------------------------------------------------------------------------
// Phase 0: f32 -> bf16, 8 elements per thread (float4 x2 in, uint4 out).
// ---------------------------------------------------------------------------
__global__ __launch_bounds__(256)
void cvt_kernel(const float* __restrict__ src, unsigned short* __restrict__ dst,
                int n) {
  const int i = (blockIdx.x * 256 + threadIdx.x) * 8;
  if (i + 8 > n) return;
  const float4 a = *reinterpret_cast<const float4*>(src + i);
  const float4 b = *reinterpret_cast<const float4*>(src + i + 4);
  uint4 u;
  u.x = pk2bf(a.x, a.y);
  u.y = pk2bf(a.z, a.w);
  u.z = pk2bf(b.x, b.y);
  u.w = pk2bf(b.z, b.w);
  *reinterpret_cast<uint4*>(dst + i) = u;
}

// ---------------------------------------------------------------------------
// Phase 1: out[b,h,blk,row,d] = bf16( X[m,:] . W[n,:] + bias[n] )
// All-bf16 GEMM. Workgroup: 128 threads (4 waves); block tile 128(M) x 64(N);
// each wave: 32 rows x 64 cols = 8 WMMAs per 32-deep k-step.
// ---------------------------------------------------------------------------
__global__ __launch_bounds__(128)
void proj_kernel(const unsigned short* __restrict__ Xb,
                 const unsigned short* __restrict__ Wb,
                 const float* __restrict__ bias,
                 unsigned short* __restrict__ out) {
  const int lane   = threadIdx.x & 31;
  const int wave   = threadIdx.x >> 5;
  const int lanelo = lane & 15;
  const int hi     = lane >> 4;
  const int mbase  = blockIdx.x * 128 + wave * 32;
  const int nbase  = blockIdx.y * 64;

  v8f acc[2][4] = {};

  for (int kb = 0; kb < H_; kb += 32) {
    // Two A fragments (rows 0-15, 16-31): k = kb + (v/4)*16 + hi*8 + (v%4)*2 + s
    v16u a[2];
#pragma unroll
    for (int mt = 0; mt < 2; ++mt) {
#pragma unroll
      for (int v = 0; v < 8; ++v) {
        const int k = kb + ((v >> 2) << 4) + (hi << 3) + ((v & 3) << 1);
        const unsigned int u = *reinterpret_cast<const unsigned int*>(
            Xb + (size_t)(mbase + mt * 16 + lanelo) * H_ + k);
        a[mt][2 * v]     = (unsigned short)(u & 0xffffu);
        a[mt][2 * v + 1] = (unsigned short)(u >> 16);
      }
    }
    // Four B fragments shared by both A fragments: k = kb + (v/4)*16 + hi*8 + (v%4)*2 + s
#pragma unroll
    for (int nt = 0; nt < 4; ++nt) {
      const int n = nbase + nt * 16 + lanelo;
      v16u bfrag;
#pragma unroll
      for (int v = 0; v < 8; ++v) {
        const int k = kb + ((v >> 2) << 4) + (hi << 3) + ((v & 3) << 1);
        const unsigned int u = *reinterpret_cast<const unsigned int*>(
            Wb + (size_t)n * H_ + k);
        bfrag[2 * v]     = (unsigned short)(u & 0xffffu);
        bfrag[2 * v + 1] = (unsigned short)(u >> 16);
      }
      acc[0][nt] = wmma_bf16(a[0], bfrag, acc[0][nt]);
      acc[1][nt] = wmma_bf16(a[1], bfrag, acc[1][nt]);
    }
  }

  // Epilogue: +bias, convert, scatter into blocked [B,NH,L,BS,HD] bf16 layout.
#pragma unroll
  for (int nt = 0; nt < 4; ++nt) {
    const int n  = nbase + nt * 16 + lanelo;
    const float bn = bias[n];
    const int h = n >> 6, d = n & 63;
#pragma unroll
    for (int mt = 0; mt < 2; ++mt) {
#pragma unroll
      for (int r = 0; r < 8; ++r) {
        const int m   = mbase + mt * 16 + (hi << 3) + r;
        const int b   = m >> 12;              // / S_
        const int s   = m & (S_ - 1);
        const int blk = s >> 6, row = s & 63;
        const size_t o =
            (((size_t)(b * NH_ + h) * L_ + blk) * BS_ + row) * HD_ + d;
        out[o] = f2bf(acc[mt][nt][r] + bn);
      }
    }
  }
}

// ---------------------------------------------------------------------------
// Phase 2: flash attention over a key-block list. One workgroup per
// (b, h, qblock); 4 waves x 16 query rows. Per key block (64 keys):
// cooperative K/V tile staging into LDS, 8 WMMA for QK^T, online softmax
// in registers, 8 WMMA for P*V.
// ---------------------------------------------------------------------------
__global__ __launch_bounds__(128)
void attn_kernel(const unsigned short* __restrict__ qbm,
                 const unsigned short* __restrict__ kbm,
                 const unsigned short* __restrict__ vbm,
                 const float* __restrict__ mask,
                 const int*   __restrict__ ra,
                 float* __restrict__ out) {
  __shared__ int kbl[L_];
  __shared__ int nkb_s;
  __shared__ unsigned short k_lds[BS_ * LDP_];        // 9.0 KB, [key][d]
  __shared__ unsigned short v_lds[BS_ * LDP_];        // 9.0 KB, [key][d]
  __shared__ unsigned short p_lds[4][16][64];         // per-wave P tile, 8 KB

  const int idx  = blockIdx.x;
  const int blk  = idx & (L_ - 1);
  const int h    = (idx >> 6) & (NH_ - 1);
  const int b    = idx >> 10;
  const int lane   = threadIdx.x & 31;
  const int wave   = threadIdx.x >> 5;
  const int lanelo = lane & 15;
  const int hi     = lane >> 4;

  if (threadIdx.x == 0) {
    int n = 0;
    if (blk == 0 || blk == L_ - 1) {            // dense rows: all blocks
      for (int i = 0; i < L_; ++i) kbl[n++] = i;
    } else if (blk == 1) {
      kbl[0] = 0; kbl[1] = 1; kbl[2] = 2; kbl[3] = L_ - 1; n = 4;
      for (int j = 0; j < R_; ++j) kbl[n++] = ra[(h * (L_ - 2) + 0) * R_ + j];
    } else if (blk == L_ - 2) {
      kbl[0] = 0; kbl[1] = L_ - 3; kbl[2] = L_ - 2; kbl[3] = L_ - 1; n = 4;
      for (int j = 0; j < R_; ++j)
        kbl[n++] = ra[(h * (L_ - 2) + (L_ - 3)) * R_ + j];
    } else {
      kbl[0] = 0; kbl[1] = blk - 1; kbl[2] = blk; kbl[3] = blk + 1;
      kbl[4] = L_ - 1; n = 5;
      for (int j = 0; j < R_; ++j)
        kbl[n++] = ra[(h * (L_ - 2) + (blk - 1)) * R_ + j];
    }
    nkb_s = n;
  }
  __syncthreads();
  const int nkb = nkb_s;

  const size_t bh = (size_t)(b * NH_ + h);

  // Q fragments for this wave's 16 rows (HD=64 => two 16x32 A-fragments).
  const unsigned short* qbase = qbm + ((bh * L_ + blk) * BS_ + wave * 16) * HD_;
  v16u qa[2];
#pragma unroll
  for (int half = 0; half < 2; ++half) {
#pragma unroll
    for (int v = 0; v < 8; ++v) {
      const int d = half * 32 + ((v >> 2) << 4) + (hi << 3) + ((v & 3) << 1);
      const unsigned int u = *reinterpret_cast<const unsigned int*>(
          qbase + (size_t)lanelo * HD_ + d);
      qa[half][2 * v]     = (unsigned short)(u & 0xffffu);
      qa[half][2 * v + 1] = (unsigned short)(u >> 16);
    }
  }

  float Mrow[8], lrow[8];
#pragma unroll
  for (int r = 0; r < 8; ++r) { Mrow[r] = -1e30f; lrow[r] = 0.0f; }
  v8f o[4] = {};

  const float rsd = 0.125f;                     // 1/sqrt(HD)

  // Staging coordinates: thread stages half a row of K and half a row of V.
  const int srow = threadIdx.x >> 1;
  const int scol = (threadIdx.x & 1) * 32;

  for (int ki = 0; ki < nkb; ++ki) {
    const int kb = kbl[ki];
    const unsigned short* kbase = kbm + (bh * L_ + kb) * BS_ * HD_;
    const unsigned short* vbase = vbm + (bh * L_ + kb) * BS_ * HD_;
    if (ki + 1 < nkb) {                         // L2 prefetch of next block
      const int kn = kbl[ki + 1];
      __builtin_prefetch(kbm + (bh * L_ + kn) * BS_ * HD_, 0, 1);
      __builtin_prefetch(vbm + (bh * L_ + kn) * BS_ * HD_, 0, 1);
    }

    // ---- cooperative staging: 8 KB K + 8 KB V, b128 in / b128 to LDS
    {
      const uint4* gk = reinterpret_cast<const uint4*>(kbase + srow * HD_ + scol);
      const uint4* gv = reinterpret_cast<const uint4*>(vbase + srow * HD_ + scol);
      uint4* lk = reinterpret_cast<uint4*>(&k_lds[srow * LDP_ + scol]);
      uint4* lv = reinterpret_cast<uint4*>(&v_lds[srow * LDP_ + scol]);
#pragma unroll
      for (int i = 0; i < 4; ++i) lk[i] = gk[i];
#pragma unroll
      for (int i = 0; i < 4; ++i) lv[i] = gv[i];
    }
    __syncthreads();

    // ---- scores: 4 tiles of 16q x 16keys, accumulate over HD in 2 WMMAs
    v8f sc[4];
#pragma unroll
    for (int t = 0; t < 4; ++t) {
      v8f c = {};
      const int key = t * 16 + lanelo;
#pragma unroll
      for (int half = 0; half < 2; ++half) {
        v16u bk;                                // K^T as B: (d,key), d=(v/4)*16+hi*8+(v%4)*2+s
#pragma unroll
        for (int v = 0; v < 8; ++v) {
          const int d = half * 32 + ((v >> 2) << 4) + (hi << 3) + ((v & 3) << 1);
          const unsigned int u = *reinterpret_cast<const unsigned int*>(
              &k_lds[key * LDP_ + d]);
          bk[2 * v]     = (unsigned short)(u & 0xffffu);
          bk[2 * v + 1] = (unsigned short)(u >> 16);
        }
        c = wmma_bf16(qa[half], bk, c);
      }
      sc[t] = c;
    }

    // scale + key-token mask penalty (column-wise; lane owns one key column)
#pragma unroll
    for (int t = 0; t < 4; ++t) {
      const float mk  = mask[(size_t)b * S_ + kb * BS_ + t * 16 + lanelo];
      const float pen = (1.0f - mk) * PEN_;
#pragma unroll
      for (int r = 0; r < 8; ++r) sc[t][r] = sc[t][r] * rsd + pen;
    }

    // ---- online softmax (rows fixed per lane; reduce across 16-lane halves)
    float alpha[8];
#pragma unroll
    for (int r = 0; r < 8; ++r) {
      float v0 = fmaxf(fmaxf(sc[0][r], sc[1][r]), fmaxf(sc[2][r], sc[3][r]));
#pragma unroll
      for (int off = 1; off < 16; off <<= 1)
        v0 = fmaxf(v0, __shfl_xor(v0, off, 16));
      const float mn = fmaxf(Mrow[r], v0);
      alpha[r] = __expf(Mrow[r] - mn);
      Mrow[r]  = mn;
    }
    float cs[8];
#pragma unroll
    for (int r = 0; r < 8; ++r) cs[r] = 0.0f;
#pragma unroll
    for (int t = 0; t < 4; ++t) {
#pragma unroll
      for (int r = 0; r < 8; ++r) {
        const float p = __expf(sc[t][r] - Mrow[r]);
        sc[t][r] = p;
        cs[r] += p;
      }
    }
#pragma unroll
    for (int r = 0; r < 8; ++r) {
      float v0 = cs[r];
#pragma unroll
      for (int off = 1; off < 16; off <<= 1) v0 += __shfl_xor(v0, off, 16);
      lrow[r] = lrow[r] * alpha[r] + v0;
    }
#pragma unroll
    for (int t = 0; t < 4; ++t)
#pragma unroll
      for (int r = 0; r < 8; ++r) o[t][r] *= alpha[r];

    // ---- P -> LDS (bf16) to re-shape into A-fragments (wave-private tile;
    //      DS ops from one wave complete in order, barrier kept for safety)
#pragma unroll
    for (int t = 0; t < 4; ++t) {
      const int col = t * 16 + lanelo;
#pragma unroll
      for (int r = 0; r < 8; ++r)
        p_lds[wave][(hi << 3) + r][col] = f2bf(sc[t][r]);
    }
    __syncthreads();

    v16u pa[2];
#pragma unroll
    for (int half = 0; half < 2; ++half) {
#pragma unroll
      for (int v = 0; v < 8; ++v) {
        const int kk = half * 32 + ((v >> 2) << 4) + (hi << 3) + ((v & 3) << 1);
        const unsigned int u = *reinterpret_cast<const unsigned int*>(
            &p_lds[wave][lanelo][kk]);
        pa[half][2 * v]     = (unsigned short)(u & 0xffffu);
        pa[half][2 * v + 1] = (unsigned short)(u >> 16);
      }
    }

    // ---- O += P * V   (V as B: (key,d), key = (v/4)*16+hi*8+(v%4)*2+s per 32-key half)
#pragma unroll
    for (int dt = 0; dt < 4; ++dt) {
      const int d = dt * 16 + lanelo;
#pragma unroll
      for (int half = 0; half < 2; ++half) {
        v16u bv;
#pragma unroll
        for (int v = 0; v < 8; ++v) {
          const int k0 = half * 32 + ((v >> 2) << 4) + (hi << 3) + ((v & 3) << 1);
          bv[2 * v]     = v_lds[k0 * LDP_ + d];
          bv[2 * v + 1] = v_lds[(k0 + 1) * LDP_ + d];
        }
        o[dt] = wmma_bf16(pa[half], bv, o[dt]);
      }
    }
    __syncthreads();   // all waves done with k_lds/v_lds before re-staging
  }

  // ---- epilogue: 1/l, query mask, write fp32 out[b, qtok, h*HD + d]
#pragma unroll
  for (int r = 0; r < 8; ++r) {
    const int m    = wave * 16 + (hi << 3) + r;
    const int qtok = blk * BS_ + m;
    const float qm = mask[(size_t)b * S_ + qtok];
    const float inv = qm / lrow[r];
    float* orow = out + ((size_t)b * S_ + qtok) * H_ + h * HD_;
#pragma unroll
    for (int dt = 0; dt < 4; ++dt)
      orow[dt * 16 + lanelo] = o[dt][r] * inv;
  }
}

// ---------------------------------------------------------------------------
extern "C" void kernel_launch(void* const* d_in, const int* in_sizes, int n_in,
                              void* d_out, int out_size, void* d_ws, size_t ws_size,
                              hipStream_t stream) {
  (void)in_sizes; (void)n_in; (void)out_size; (void)ws_size;
  const float* X[3]  = {(const float*)d_in[0], (const float*)d_in[1],
                        (const float*)d_in[2]};                  // Q, K, V
  const float* mask  = (const float*)d_in[3];
  const float* W[3]  = {(const float*)d_in[4], (const float*)d_in[6],
                        (const float*)d_in[8]};                  // Wq, Wk, Wv
  const float* bias[3] = {(const float*)d_in[5], (const float*)d_in[7],
                          (const float*)d_in[9]};                // bq, bk, bv
  const int* ra = (const int*)d_in[10];
  float* out = (float*)d_out;

  const size_t NE = (size_t)B_ * NH_ * L_ * BS_ * HD_;   // 8,388,608 elems
  const size_t NW = (size_t)H_ * H_;                      // 1,048,576 elems
  unsigned short* blkout[3];                              // q/k/v blocked bf16
  blkout[0] = (unsigned short*)d_ws;
  blkout[1] = blkout[0] + NE;
  blkout[2] = blkout[1] + NE;
  unsigned short* xbuf = blkout[2] + NE;                  // reused per proj
  unsigned short* wbuf = xbuf + NE;                       // reused per proj

  const dim3 pgrid((B_ * S_) / 128, H_ / 64, 1);          // 64 x 16
  for (int p = 0; p < 3; ++p) {
    cvt_kernel<<<(int)(NE / (256 * 8)), 256, 0, stream>>>(X[p], xbuf, (int)NE);
    cvt_kernel<<<(int)(NW / (256 * 8)), 256, 0, stream>>>(W[p], wbuf, (int)NW);
    proj_kernel<<<pgrid, 128, 0, stream>>>(xbuf, wbuf, bias[p], blkout[p]);
  }

  attn_kernel<<<B_ * NH_ * L_, 128, 0, stream>>>(blkout[0], blkout[1],
                                                 blkout[2], mask, ra, out);
}
